// Model_21706764714353
// MI455X (gfx1250) — matrix operand order repaired and checked by reading, not compiled, not run on hardware
//
#include <hip/hip_runtime.h>

typedef __attribute__((ext_vector_type(16))) _Float16 v16h;
typedef __attribute__((ext_vector_type(8)))  _Float16 v8h;
typedef __attribute__((ext_vector_type(8)))  float    v8f;

#define B_TOT 256
#define L_TOT 336
#define C_TOT 862
#define SEG   56     // segments (K of GEMM1); also P range of GEMM2 output
#define WIN   6
#define HID   16
#define P_OUT 56
#define BT    2      // batches per workgroup
#define CT    32     // channels per workgroup
#define NW    8      // waves per workgroup (256 threads, wave32)
#define CPW   4      // channels per wave
#define XSTR  72     // ldsX row stride in halves: 144B (16B aligned, bank str 36)
#define NROW  (BT*WIN*CT)                 // 384 (b,win,channel) rows

#define X_BYTES  ((size_t)NROW * XSTR * 2)        // 55296
#define O_BYTES  ((size_t)NROW * P_OUT * 4)       // 86016
#define M_BYTES  ((size_t)BT * CT * 4)            // 256
#define H_BYTES  ((size_t)NW * 256 * 2)           // 4096
#define BB_BYTES ((size_t)NW * P_OUT * 4)         // 1792
#define SMEM_BYTES (X_BYTES + O_BYTES + M_BYTES + H_BYTES + BB_BYTES)

#define WS1_ELEMS  ((size_t)C_TOT * 2 * 32 * 16)  // A1 frags: [c][kc][lane][16] f16
#define WS2_ELEMS  ((size_t)C_TOT * 4 * 32 * 16)  // A2 frags: [c][mt][lane][16] f16
#define WSB_ELEMS  ((size_t)C_TOT * P_OUT)        // beta[c][p] f32
#define WS_BYTES   (WS1_ELEMS*2 + WS2_ELEMS*2 + WSB_ELEMS*4)

__device__ __forceinline__ int imin(int a, int b) { return a < b ? a : b; }

// e[n] = sum_k D[k,n] for the 6x6 ortho DCT-II basis (precomputed)
__constant__ float E6[6] = {2.3122790f, -0.5773503f, 0.4957819f,
                            -0.1019354f, 0.2391412f, 0.0815680f};

// ---------------------------------------------------------------------------
// Prep kernel: pack W1/W2 into CDNA5 16-bit A-fragment lane layout (f16),
// and precompute beta[c][p] = b2[c,p] + sum_h b1[c,h]*W2[c,h,p].
// ---------------------------------------------------------------------------
__global__ __launch_bounds__(256) void pack_weights_kernel(
    const float* __restrict__ W1, const float* __restrict__ b1,
    const float* __restrict__ W2, const float* __restrict__ b2,
    _Float16* __restrict__ ws1, _Float16* __restrict__ ws2,
    float* __restrict__ wsb)
{
  const int c   = blockIdx.x;
  const int tid = threadIdx.x;
  const float* w1c = W1 + (size_t)c * (SEG * HID);
  const float* w2c = W2 + (size_t)c * (HID * P_OUT);

  // GEMM1 A fragments (W1^T, M=h, K=s, K padded 56->64): idx=(kc*32+lane)*16+j
  for (int i = tid; i < 2 * 32 * 16; i += 256) {
    int j = i & 15, lane = (i >> 4) & 31, kc = i >> 9;
    int hi = (lane >> 4) & 1, m = lane & 15;
    int s = kc * 32 + ((j < 8) ? j : (j + 8)) + hi * 8;
    float w = w1c[imin(s, SEG - 1) * HID + m];
    ws1[(size_t)c * 1024 + i] = (s < SEG) ? (_Float16)w : (_Float16)0.f;
  }
  // GEMM2 A fragments (W2^T, M=p padded 56->64, K=h padded 16->32)
  for (int i = tid; i < 4 * 32 * 16; i += 256) {
    int j = i & 15, lane = (i >> 4) & 31, mt = i >> 9;
    int hi = (lane >> 4) & 1, p = mt * 16 + (lane & 15);
    int k = ((j < 8) ? j : (j + 8)) + hi * 8;
    float w = w2c[imin(k, HID - 1) * P_OUT + imin(p, P_OUT - 1)];
    ws2[(size_t)c * 2048 + i] =
        (k < HID && p < P_OUT) ? (_Float16)w : (_Float16)0.f;
  }
  // beta
  if (tid < P_OUT) {
    float bv = b2[(size_t)c * P_OUT + tid];
    #pragma unroll
    for (int h = 0; h < HID; ++h)
      bv += b1[(size_t)c * HID + h] * w2c[h * P_OUT + tid];
    wsb[(size_t)c * P_OUT + tid] = bv;
  }
}

// ---------------------------------------------------------------------------
// Shared device body. PREPACKED=1 reads fragment-ordered f16 weights + beta
// from workspace; PREPACKED=0 is the self-contained fallback.
// LDS x layout: row = (b*WIN + win)*CT + cc, contiguous in s (padded 56->64,
// pad slots zeroed) -> GEMM1 B fragments are two ds_load_b128, zero VALU.
// ---------------------------------------------------------------------------
template <int PREPACKED>
__device__ __forceinline__ void sparsetsf_body(
    const float* __restrict__ x,  const float* __restrict__ W1,
    const float* __restrict__ b1, const float* __restrict__ W2,
    const float* __restrict__ b2, float* __restrict__ out,
    const _Float16* __restrict__ ws1, const _Float16* __restrict__ ws2,
    const float* __restrict__ wsb)
{
  extern __shared__ char smem[];
  _Float16* ldsX   = (_Float16*)smem;                           // [NROW][XSTR]
  float*    ldsO   = (float*)(smem + X_BYTES);                  // [NROW][P_OUT]
  float*    ldsMean= (float*)(smem + X_BYTES + O_BYTES);        // [BT*CT]
  _Float16* ldsHT  = (_Float16*)(smem + X_BYTES + O_BYTES + M_BYTES); // [NW][256]
  float*    ldsBeta= (float*)(smem + X_BYTES + O_BYTES + M_BYTES + H_BYTES);

  const int c0   = blockIdx.x * CT;
  const int b0   = blockIdx.y * BT;
  const int tid  = threadIdx.x;
  const int lane = tid & 31;
  const int wave = tid >> 5;
  const int nlane = lane & 15;
  const int hi    = (lane >> 4) & 1;        // 0: lanes 0-15, 1: lanes 16-31
  const bool fullc = (c0 + CT <= C_TOT);    // uniform: no boundary channels

  // ---- Phase 1: coalesced global -> LDS (transposed to s-contiguous rows) --
  if (fullc) {
    for (int i = tid; i < BT*L_TOT*CT; i += 256) {
      int cc  = i & (CT - 1);
      int row = i >> 5;                     // b*L + l
      int b   = row / L_TOT;
      int l   = row - b * L_TOT;
      int s   = l / WIN, win = l - s * WIN;
      float v = x[((size_t)(b0 + b) * L_TOT + l) * C_TOT + (c0 + cc)];
      ldsX[((b * WIN + win) * CT + cc) * XSTR + s] = (_Float16)v;
    }
  } else {
    for (int i = tid; i < BT*L_TOT*CT; i += 256) {
      int cc  = i & (CT - 1);
      int row = i >> 5;
      int b   = row / L_TOT;
      int l   = row - b * L_TOT;
      int s   = l / WIN, win = l - s * WIN;
      int c   = imin(c0 + cc, C_TOT - 1);   // clamp: garbage lanes never consumed
      float v = x[((size_t)(b0 + b) * L_TOT + l) * C_TOT + c];
      ldsX[((b * WIN + win) * CT + cc) * XSTR + s] = (_Float16)v;
    }
  }
  // zero the K-padding slots s = 56..63 of every row
  for (int i = tid; i < NROW * 8; i += 256) {
    int row = i >> 3, s = SEG + (i & 7);
    ldsX[row * XSTR + s] = (_Float16)0.f;
  }
  __syncthreads();

  // ---- Phase 2: per-(batch,channel) mean ----
  if (tid < BT * CT) {
    int b = tid >> 5, cc = tid & 31;
    float sum = 0.f;
    for (int w = 0; w < WIN; ++w) {
      const _Float16* p = ldsX + ((b * WIN + w) * CT + cc) * XSTR;
      for (int s = 0; s < SEG; ++s) sum += (float)p[s];
    }
    ldsMean[tid] = sum * (1.0f / L_TOT);
  }
  __syncthreads();

  // ---- Phase 3: subtract mean in place (pad slots untouched, stay zero) ----
  for (int i = tid; i < NROW * SEG; i += 256) {
    int row = i / SEG, s = i - row * SEG;
    int b   = row / (WIN * CT);
    int cc  = row & (CT - 1);
    float m = ldsMean[b * CT + cc];
    int idx = row * XSTR + s;
    ldsX[idx] = (_Float16)((float)ldsX[idx] - m);
  }
  __syncthreads();

  // ---- Phase 4: per-wave WMMA pipeline over this wave's 4 channels ----
  const int  col   = nlane;                     // col = b*WIN + win (12/16 used)
  const bool colok = (col < BT * WIN);
  const int  bb    = imin(col / WIN, BT - 1);   // clamped for safe addressing
  const int  win   = col % WIN;

  for (int ci = 0; ci < CPW; ++ci) {
    const int cc = wave * CPW + ci;
    const int c  = c0 + cc;
    if (c >= C_TOT) continue;                   // wave-uniform: safe
    const float* w1c = W1 + (size_t)c * (SEG * HID);
    const float* w2c = W2 + (size_t)c * (HID * P_OUT);
    const _Float16* xrow = ldsX + ((bb * WIN + win) * CT + cc) * XSTR;

    // ---- GEMM1: H[16 x cols] = W1^T (16x56) * Xc (56 x 12), K padded to 64
    v8f hacc;
    #pragma unroll
    for (int r = 0; r < 8; ++r) hacc[r] = 0.f;

    #pragma unroll
    for (int kc = 0; kc < 2; ++kc) {
      v16h a;
      if (PREPACKED) {
        a = *(const v16h*)(ws1 + (size_t)c * 1024 + (kc * 32 + lane) * 16);
      } else {
        #pragma unroll
        for (int j = 0; j < 16; ++j) {
          int s = kc * 32 + ((j < 8) ? j : (j + 8)) + hi * 8;
          float w = w1c[imin(s, SEG - 1) * HID + nlane];
          a[j] = (s < SEG) ? (_Float16)w : (_Float16)0.f;
        }
      }
      // B fragment: half j<8 -> K = kc*32 + hi*8 + j, j>=8 -> K = kc*32 + 16 + hi*8 + (j-8).
      // Cols >=12 read valid garbage rows; their output is discarded.
      const _Float16* bp = xrow + kc * 32 + hi * 8;
      v8h blo = *(const v8h*)bp;
      v8h bhi = *(const v8h*)(bp + 16);
      v16h bf;
      #pragma unroll
      for (int j = 0; j < 8; ++j) { bf[j] = blo[j]; bf[j + 8] = bhi[j]; }
      hacc = __builtin_amdgcn_wmma_f32_16x16x32_f16(
          false, a, false, bf, (short)0, hacc, false, false);
    }

    // Stage H transposed as f16: hWT[n*16 + m]; lane's 8 values contiguous.
    _Float16* hWT = ldsHT + wave * 256;
    v8h hst;
    #pragma unroll
    for (int r = 0; r < 8; ++r) hst[r] = (_Float16)hacc[r];
    *(v8h*)(hWT + nlane * 16 + hi * 8) = hst;

    // beta_c[p] into per-wave LDS (unconditional: clamped lanes rewrite p=55)
    if (PREPACKED) {
      int p2c = imin(lane + 32, P_OUT - 1);
      ldsBeta[wave * P_OUT + lane] = wsb[(size_t)c * P_OUT + lane];
      ldsBeta[wave * P_OUT + p2c]  = wsb[(size_t)c * P_OUT + p2c];
    } else {
      int p2c = imin(lane + 32, P_OUT - 1);
      float bv  = b2[(size_t)c * P_OUT + lane];
      float bv2 = b2[(size_t)c * P_OUT + p2c];
      #pragma unroll
      for (int h = 0; h < HID; ++h) {
        float b1v = b1[(size_t)c * HID + h];
        bv  += b1v * w2c[h * P_OUT + lane];
        bv2 += b1v * w2c[h * P_OUT + p2c];
      }
      ldsBeta[wave * P_OUT + lane] = bv;
      ldsBeta[wave * P_OUT + p2c]  = bv2;
    }

    // B fragment for GEMM2: K = h (slot j<8 -> h = hi*8 + j); slots j>=8 supply the zero padding
    // (explicit zero: first iteration reads uninitialized LDS, avoid NaN*0)
    v8h h0 = *(const v8h*)(hWT + nlane * 16);
    v8h h1 = *(const v8h*)(hWT + nlane * 16 + 8);
    v16h b2f;
    #pragma unroll
    for (int j = 0; j < 8; ++j) {
      b2f[j]     = hi ? h1[j] : h0[j];
      b2f[j + 8] = (_Float16)0.f;
    }

    // ---- GEMM2: Y[56 x cols] = W2^T (56x16) * H, 4 M-tiles, K padded to 32
    const float mval = ldsMean[bb * CT + cc];
    const float ev   = E6[win];
    float* orow = ldsO + ((bb * WIN + win) * CT + cc) * P_OUT;
    #pragma unroll
    for (int mt = 0; mt < 4; ++mt) {
      v16h a2;
      if (PREPACKED) {
        a2 = *(const v16h*)(ws2 + (size_t)c * 2048 + (mt * 32 + lane) * 16);
      } else {
        int p = mt * 16 + nlane;
        #pragma unroll
        for (int j = 0; j < 16; ++j) {
          int k = ((j < 8) ? j : (j + 8)) + hi * 8;       // h index, >=16 is pad
          float w = w2c[imin(k, HID - 1) * P_OUT + imin(p, P_OUT - 1)];
          a2[j] = (k < HID && p < P_OUT) ? (_Float16)w : (_Float16)0.f;
        }
      }
      v8f y;
      #pragma unroll
      for (int r = 0; r < 8; ++r) y[r] = 0.f;
      y = __builtin_amdgcn_wmma_f32_16x16x32_f16(
          false, a2, false, b2f, (short)0, y, false, false);

      // Epilogue: element (pp = mt*16 + r + hi*8, col); pp consecutive in r,
      // so stores merge into wide ds stores.
      if (colok) {
        #pragma unroll
        for (int r = 0; r < 8; ++r) {
          int pp = mt * 16 + r + hi * 8;
          if (pp < P_OUT)
            orow[pp] = y[r] + ev * ldsBeta[wave * P_OUT + pp] + mval;
        }
      }
    }
  }
  __syncthreads();

  // ---- Phase 5: coalesced LDS -> global writeback ----
  if (fullc) {
    for (int i = tid; i < BT*L_TOT*CT; i += 256) {
      int cc  = i & 31;
      int row = i >> 5;
      int b   = row / L_TOT;
      int l   = row - b * L_TOT;
      int p   = l / WIN, w = l - p * WIN;
      out[((size_t)(b0 + b) * L_TOT + l) * C_TOT + (c0 + cc)] =
          ldsO[((b * WIN + w) * CT + cc) * P_OUT + p];
    }
  } else {
    for (int i = tid; i < BT*L_TOT*CT; i += 256) {
      int cc  = i & 31;
      int row = i >> 5;
      int b   = row / L_TOT;
      int l   = row - b * L_TOT;
      int p   = l / WIN, w = l - p * WIN;
      int c   = c0 + cc;
      if (c < C_TOT)
        out[((size_t)(b0 + b) * L_TOT + l) * C_TOT + c] =
            ldsO[((b * WIN + w) * CT + cc) * P_OUT + p];
    }
  }
}

__global__ __launch_bounds__(256) void sparsetsf_wmma_packed(
    const float* __restrict__ x, float* __restrict__ out,
    const _Float16* __restrict__ ws1, const _Float16* __restrict__ ws2,
    const float* __restrict__ wsb)
{
  sparsetsf_body<1>(x, nullptr, nullptr, nullptr, nullptr, out, ws1, ws2, wsb);
}

__global__ __launch_bounds__(256) void sparsetsf_wmma_fallback(
    const float* __restrict__ x,  const float* __restrict__ W1,
    const float* __restrict__ b1, const float* __restrict__ W2,
    const float* __restrict__ b2, float* __restrict__ out)
{
  sparsetsf_body<0>(x, W1, b1, W2, b2, out, nullptr, nullptr, nullptr);
}

extern "C" void kernel_launch(void* const* d_in, const int* in_sizes, int n_in,
                              void* d_out, int out_size, void* d_ws, size_t ws_size,
                              hipStream_t stream) {
  (void)in_sizes; (void)n_in; (void)out_size;
  const float* x  = (const float*)d_in[0];
  const float* W1 = (const float*)d_in[1];
  const float* b1 = (const float*)d_in[2];
  const float* W2 = (const float*)d_in[3];
  const float* b2 = (const float*)d_in[4];
  float* out = (float*)d_out;

  dim3 grid((C_TOT + CT - 1) / CT, B_TOT / BT);
  dim3 block(256);

  if (ws_size >= WS_BYTES) {
    _Float16* ws1 = (_Float16*)d_ws;
    _Float16* ws2 = ws1 + WS1_ELEMS;
    float*    wsb = (float*)(ws2 + WS2_ELEMS);
    pack_weights_kernel<<<dim3(C_TOT), block, 0, stream>>>(W1, b1, W2, b2,
                                                           ws1, ws2, wsb);
    sparsetsf_wmma_packed<<<grid, block, SMEM_BYTES, stream>>>(x, out,
                                                               ws1, ws2, wsb);
  } else {
    sparsetsf_wmma_fallback<<<grid, block, SMEM_BYTES, stream>>>(x, W1, b1,
                                                                 W2, b2, out);
  }
}
